// LbpBlock1_31636729102829
// MI455X (gfx1250) — matrix operand order repaired and checked by reading, not compiled, not run on hardware
//
#include <hip/hip_runtime.h>

typedef __attribute__((ext_vector_type(16))) _Float16 v16h;
typedef __attribute__((ext_vector_type(8)))  float    v8f;

#define IMG   256
#define INNER 254
#define NIMG  32
#define OCH   64   // matmul output channels
#define TCH   67   // total output channels (3 copy + 64)

// ---------------------------------------------------------------------------
// Kernel 1: fold the two einsum weight tensors into W2[k][o], k = i*8+l (24
// used, padded to 32 with zeros), o = 0..63.  W2 is tiny (8 KB) -> d_ws.
// ---------------------------------------------------------------------------
__global__ void prep_weights(const float* __restrict__ w,   // (3,5,8)
                             const float* __restrict__ wl,  // (5,64)
                             float* __restrict__ W2) {      // (32,64)
  for (int idx = threadIdx.x; idx < 32 * OCH; idx += blockDim.x) {
    int k = idx >> 6, o = idx & 63;
    float acc = 0.f;
    if (k < 24) {
      int i = k >> 3, l = k & 7;
      #pragma unroll
      for (int m = 0; m < 5; ++m)
        acc += expf(w[i * 40 + m * 8 + l]) * wl[m * 64 + o];
    }
    W2[idx] = acc;  // rows 24..31 stay exactly zero (K padding)
  }
}

// ---------------------------------------------------------------------------
// Kernel 2: channels 0..2 (zero-bordered copy of x) + zero frame of channels
// 3..66.  Must run every launch: d_out is poisoned before timing.
// ---------------------------------------------------------------------------
__global__ void copy_border(const float* __restrict__ x, float* __restrict__ out) {
  const long long N1 = (long long)NIMG * 3 * IMG * IMG;      // copy channels
  const long long N2 = (long long)NIMG * OCH * 1020;         // frame pixels
  long long idx = (long long)blockIdx.x * blockDim.x + threadIdx.x;
  if (idx < N1) {
    int pix = (int)(idx & 65535);
    int plane = (int)(idx >> 16);           // n*3 + ch
    int n = plane / 3, ch = plane % 3;
    int r = pix >> 8, c = pix & 255;
    bool interior = (r >= 1) & (r <= INNER) & (c >= 1) & (c <= INNER);
    out[(((long long)n * TCH + ch) << 16) + pix] = interior ? x[idx] : 0.f;
  } else if (idx < N1 + N2) {
    long long j = idx - N1;
    int b = (int)(j % 1020);
    int no_ = (int)(j / 1020);
    int n = no_ >> 6, o = no_ & 63;
    int r, c;
    if (b < 256)      { r = 0;        c = b;       }
    else if (b < 512) { r = 255;      c = b - 256; }
    else if (b < 766) { r = b - 511;  c = 0;       }
    else              { r = b - 765;  c = 255;     }
    out[(((long long)n * TCH + 3 + o) << 16) + r * 256 + c] = 0.f;
  }
}

// ---------------------------------------------------------------------------
// Kernel 3: per-pixel 24x64 matmul via v_wmma_f32_16x16x32_f16.
//   A (SRC0) = weights, M = 16 output channels, K = 32 (hi/lo f16 split)
//   B (SRC1) = sign diffs, K = 32, N = 16 consecutive pixel columns
//   C lanes index pixel columns -> coalesced stores.
// One wave per (image, row); 16 column-tiles per row; 8 WMMAs per tile.
// ---------------------------------------------------------------------------
__global__ __launch_bounds__(256)
void lbp_wmma(const float* __restrict__ x, const float* __restrict__ W2,
              float* __restrict__ out) {
  const int lane    = threadIdx.x & 31;
  const int nl      = lane & 15;
  const bool hihalf = lane >= 16;
  const int wave    = blockIdx.x * (blockDim.x >> 5) + (threadIdx.x >> 5);
  const int n   = wave / INNER;          // image
  const int row = 1 + wave % INNER;      // interior row 1..254

  // ---- A fragments: weights, hi/lo f16 split (signs are exact in f16, so
  // A_hi*B + A_lo*B reconstructs fp32-grade accuracy in the f32 accumulator).
  // A fragment in this code's contiguous k order: lanes 0-15 hold M=lane, K=0..15 (e0..15);
  // lanes 16-31 hold M=lane-16, K=16..31.
  v16h ahi[4], alo[4];
  #pragma unroll
  for (int t = 0; t < 4; ++t) {
    #pragma unroll
    for (int e = 0; e < 16; ++e) {
      int k = (hihalf ? 16 : 0) + e;
      float v = W2[k * OCH + t * 16 + nl];
      _Float16 h = (_Float16)v;
      ahi[t][e] = h;
      alo[t][e] = (_Float16)(v - (float)h);
    }
  }

  const float* xn = x + (long long)n * 3 * IMG * IMG;
  float* outb = out + ((long long)n * TCH + 3) * IMG * IMG + (long long)row * IMG;

  // LBP neighbor offsets in reference _OFFSETS order (row stride = 256)
  const int noff[8] = {-257, -256, -255, -1, 255, 256, 257, 1};

  for (int t16 = 0; t16 < 16; ++t16) {
    const int col0 = 1 + 16 * t16;
    const int col  = col0 + nl;                  // this lane's pixel column
    const int colc = col > INNER ? INNER : col;  // clamp tail-tile addresses

    // ---- B fragment: sign diffs.  same contiguous k order as A: lanes 0-15 hold
    // N=lane, K=0..15 (ch0 offs + ch1 offs); lanes 16-31 hold N=lane-16,
    // K=16..31 (ch2 offs + zero pad).
    v16h b;
    {
      const int ia = hihalf ? 2 : 0;
      const float* bp = xn + ((long long)ia * IMG + row) * IMG + colc;
      float c0 = bp[0];
      #pragma unroll
      for (int q = 0; q < 8; ++q) {
        float nb = bp[noff[q]];
        b[q] = (_Float16)(float)((c0 > nb) - (c0 < nb));
      }
      if (!hihalf) {
        const float* bp1 = xn + ((long long)1 * IMG + row) * IMG + colc;
        float c1 = bp1[0];
        #pragma unroll
        for (int q = 0; q < 8; ++q) {
          float nb = bp1[noff[q]];
          b[8 + q] = (_Float16)(float)((c1 > nb) - (c1 < nb));
        }
      } else {
        #pragma unroll
        for (int q = 0; q < 8; ++q) b[8 + q] = (_Float16)0.f;  // K 24..31 pad
      }
    }

    // ---- 8 WMMAs: 4 channel-tiles x (hi + lo)
    v8f acc[4] = {};
    #pragma unroll
    for (int t = 0; t < 4; ++t) {
      acc[t] = __builtin_amdgcn_wmma_f32_16x16x32_f16(
          false, ahi[t], false, b, (short)0, acc[t], false, false);
      acc[t] = __builtin_amdgcn_wmma_f32_16x16x32_f16(
          false, alo[t], false, b, (short)0, acc[t], false, false);
    }

    // ---- Stores.  C 16x16 f32 layout: lane gives N (=pixel col), VGPR r
    // gives M=r (lanes 0-15) / r+8 (lanes 16-31).  Each store instruction
    // writes 2 channels x 16 consecutive columns -> coalesced.
    if (col <= INNER) {
      #pragma unroll
      for (int t = 0; t < 4; ++t) {
        const int chm = t * 16 + (hihalf ? 8 : 0);
        #pragma unroll
        for (int r = 0; r < 8; ++r)
          outb[(long long)(chm + r) * IMG * IMG + col] = acc[t][r];
      }
    }
  }
}

extern "C" void kernel_launch(void* const* d_in, const int* in_sizes, int n_in,
                              void* d_out, int out_size, void* d_ws, size_t ws_size,
                              hipStream_t stream) {
  const float* x  = (const float*)d_in[0];   // (32,3,256,256)
  const float* w  = (const float*)d_in[1];   // (3,5,8)
  const float* wl = (const float*)d_in[2];   // (5,64)
  float* out = (float*)d_out;                // (32,67,256,256)
  float* W2  = (float*)d_ws;                 // 32*64 floats = 8 KB

  prep_weights<<<1, 256, 0, stream>>>(w, wl, W2);

  const long long total_cb = (long long)NIMG * 3 * IMG * IMG
                           + (long long)NIMG * OCH * 1020;
  copy_border<<<(int)((total_cb + 255) / 256), 256, 0, stream>>>(x, out);

  // one wave per (image,row): 32*254 = 8128 waves; 8 waves (256 thr) / block
  lbp_wmma<<<(NIMG * INNER) / 8, 256, 0, stream>>>(x, W2, out);
}
